// SelfAttention_44538810859826
// MI455X (gfx1250) — matrix operand order repaired and checked by reading, not compiled, not run on hardware
//
#include <hip/hip_runtime.h>
#include <hip/hip_bf16.h>

typedef _Float16 half_t;
typedef __attribute__((ext_vector_type(16))) _Float16 v16h;
typedef __attribute__((ext_vector_type(8)))  _Float16 v8h;
typedef __attribute__((ext_vector_type(8)))  float    v8f;

// ---------------------------------------------------------------------------
// helpers
// ---------------------------------------------------------------------------
__device__ __forceinline__ v16h ldfrag(const half_t* p0, const half_t* p1) {
    v8h lo = *(const v8h*)p0;
    v8h hi = *(const v8h*)p1;
    return __builtin_shufflevector(lo, hi, 0,1,2,3,4,5,6,7,8,9,10,11,12,13,14,15);
}

__device__ __forceinline__ v8f wmma_f16(v16h a, v16h b, v8f c) {
    return __builtin_amdgcn_wmma_f32_16x16x32_f16(false, a, false, b, (short)0, c, false, false);
}

__device__ __forceinline__ float gelu_exact(float x) {
    return 0.5f * x * (1.0f + erff(x * 0.70710678118654752f));
}

// ---------------------------------------------------------------------------
// setup kernels
// ---------------------------------------------------------------------------
__global__ void k_f32_to_f16(const float* __restrict__ a, half_t* __restrict__ b, long n) {
    long i = (long)blockIdx.x * blockDim.x + threadIdx.x;
    if (i < n) b[i] = (half_t)a[i];
}

// biasTab[h][n][m] = rpb[relidx(n,m)*8 + h];  grid = 8*128 blocks, 128 threads
__global__ void k_build_bias(const float* __restrict__ rpb, float* __restrict__ bt) {
    int h = blockIdx.x >> 7, n = blockIdx.x & 127, m = threadIdx.x;
    int t1 = n >> 6, h1 = (n >> 3) & 7, w1 = n & 7;
    int t2 = m >> 6, h2 = (m >> 3) & 7, w2 = m & 7;
    int idx = (t1 - t2 + 1) * 225 + (h1 - h2 + 7) * 15 + (w1 - w2 + 7);
    bt[((long)h * 128 + n) * 128 + m] = rpb[idx * 8 + h];
}

// shift-mask region ids: grp[win][tok]; grid=256 blocks, 128 threads
__global__ void k_build_grp(unsigned char* __restrict__ grp) {
    int win = blockIdx.x, n = threadIdx.x;
    int wt = win >> 6, wh = (win >> 3) & 7, ww = win & 7;
    int lt = n >> 6,   lh = (n >> 3) & 7,  lw = n & 7;
    int t  = wt * 2 + lt, hh = wh * 8 + lh, wc = ww * 8 + lw;
    int rt = (t  < 6)  ? 0 : ((t  < 7)  ? 1 : 2);
    int rh = (hh < 56) ? 0 : ((hh < 60) ? 1 : 2);
    int rw = (wc < 56) ? 0 : ((wc < 60) ? 1 : 2);
    grp[win * 128 + n] = (unsigned char)(rt * 9 + rh * 3 + rw);
}

// ---------------------------------------------------------------------------
// LayerNorm (+ optional window-partition gather with roll)
// one wave (32 lanes) per token, 8 tokens per 256-thread block
// mode 0: out[token] (token order); mode 1: out[win*128+n] gathered with shift
// ---------------------------------------------------------------------------
__global__ __launch_bounds__(256) void k_layernorm(
    const float* __restrict__ x, const float* __restrict__ g,
    const float* __restrict__ be, half_t* __restrict__ out,
    int mode, int st, int sh, int sw)
{
    int wave = threadIdx.x >> 5, lane = threadIdx.x & 31;
    long dtok = (long)blockIdx.x * 8 + wave;
    long stok;
    if (mode == 0) {
        stok = dtok;
    } else {
        int win = (int)(dtok >> 7), n = (int)(dtok & 127);
        int b  = win >> 8, wrem = win & 255;
        int wt = wrem >> 6, wh = (wrem >> 3) & 7, ww = wrem & 7;
        int lt = n >> 6,    lh = (n >> 3) & 7,   lw = n & 7;
        int t  = (wt * 2 + lt + st) & 7;
        int hh = (wh * 8 + lh + sh) & 63;
        int wc = (ww * 8 + lw + sw) & 63;
        stok = (((long)b * 8 + t) * 64 + hh) * 64 + wc;
    }
    const float* xr = x + stok * 256 + lane * 8;
    float v[8];
    {
        float4 a0 = *(const float4*)xr;
        float4 a1 = *(const float4*)(xr + 4);
        v[0]=a0.x; v[1]=a0.y; v[2]=a0.z; v[3]=a0.w;
        v[4]=a1.x; v[5]=a1.y; v[6]=a1.z; v[7]=a1.w;
    }
    float s = 0.f, q = 0.f;
    #pragma unroll
    for (int j = 0; j < 8; ++j) { s += v[j]; q += v[j] * v[j]; }
    #pragma unroll
    for (int d = 16; d >= 1; d >>= 1) { s += __shfl_xor(s, d); q += __shfl_xor(q, d); }
    float mean = s * (1.0f / 256.0f);
    float var  = q * (1.0f / 256.0f) - mean * mean;
    float rstd = rsqrtf(var + 1e-5f);
    v8h ov;
    #pragma unroll
    for (int j = 0; j < 8; ++j) {
        int c = lane * 8 + j;
        ov[j] = (half_t)((v[j] - mean) * rstd * g[c] + be[c]);
    }
    *(v8h*)(out + dtok * 256 + lane * 8) = ov;
}

// ---------------------------------------------------------------------------
// Tiled WMMA GEMM:  C[M,N] = A[M,K](f16) * B[K,N](f16) + bias
// block tile 128x128, K-step 64; 8 waves, wave tile 64x32 (4x2 WMMA frags)
// EPI 0: f16 store, cols < qcols scaled by qscale (QKV)
// EPI 1: GELU -> f16 store (MLP up)
// EPI 2: fp32 residual accumulate into resid (MLP down)
// EPI 3: fp32 store (proj)
// ---------------------------------------------------------------------------
template <int EPI>
__global__ __launch_bounds__(256) void k_gemm(
    const half_t* __restrict__ A, const half_t* __restrict__ B,
    const float* __restrict__ bias, void* __restrict__ out,
    float* __restrict__ resid, int M, int Nn, int K, float qscale, int qcols)
{
    constexpr int BM = 128, BN = 128, BK = 64;
    __shared__ half_t As[BM][BK];       // 16 KB
    __shared__ half_t Bs[BN][BK + 8];   // transposed, padded stride 72 (18 KB)

    int tid  = threadIdx.x;
    int wave = tid >> 5, lane = tid & 31;
    int l16  = lane & 15, lh = lane >> 4;
    int bm = (int)blockIdx.y * BM;
    int bn = (int)blockIdx.x * BN;
    int wrow = (wave >> 2) * 64;
    int wcol = (wave & 3) * 32;

    v8f acc[4][2] = {};

    int ar = tid >> 1, ac = (tid & 1) * 32;          // A-tile role of this thread

    for (int k0 = 0; k0 < K; k0 += BK) {
        // A tile: 2 threads per row, 32 f16 each
        {
            const half_t* src = A + (size_t)(bm + ar) * K + k0 + ac;
            *(v8h*)&As[ar][ac]      = *(const v8h*)(src);
            *(v8h*)&As[ar][ac + 8]  = *(const v8h*)(src + 8);
            *(v8h*)&As[ar][ac + 16] = *(const v8h*)(src + 16);
            *(v8h*)&As[ar][ac + 24] = *(const v8h*)(src + 24);
        }
        // B tile transposed: 512 chunks of 16 contiguous f16
        #pragma unroll
        for (int i = 0; i < 2; ++i) {
            int cidx = tid + 256 * i;
            int kk = cidx >> 3;
            int n0 = (cidx & 7) * 16;
            const half_t* src = B + (size_t)(k0 + kk) * Nn + bn + n0;
            v8h x0 = *(const v8h*)src;
            v8h x1 = *(const v8h*)(src + 8);
            #pragma unroll
            for (int j = 0; j < 8; ++j) Bs[n0 + j][kk]     = x0[j];
            #pragma unroll
            for (int j = 0; j < 8; ++j) Bs[n0 + 8 + j][kk] = x1[j];
        }
        // prefetch next K-tile toward WGP while this tile computes
        if (k0 + BK < K) {
            __builtin_prefetch(A + (size_t)(bm + ar) * K + (k0 + BK) + ac, 0, 0);
            __builtin_prefetch(B + (size_t)(k0 + BK + (tid >> 3)) * Nn + bn + (tid & 7) * 16, 0, 0);
        }
        __syncthreads();

        #pragma unroll
        for (int ks = 0; ks < BK; ks += 32) {
            v16h af[4], bf[2];
            #pragma unroll
            for (int i = 0; i < 4; ++i) {
                const half_t* p = &As[wrow + i * 16 + l16][ks + lh * 8];
                af[i] = ldfrag(p, p + 16);
            }
            #pragma unroll
            for (int j = 0; j < 2; ++j) {
                const half_t* p = &Bs[wcol + j * 16 + l16][ks + lh * 8];
                bf[j] = ldfrag(p, p + 16);
            }
            #pragma unroll
            for (int i = 0; i < 4; ++i)
                #pragma unroll
                for (int j = 0; j < 2; ++j)
                    acc[i][j] = wmma_f16(af[i], bf[j], acc[i][j]);
        }
        __syncthreads();
    }

    // epilogue
    #pragma unroll
    for (int i = 0; i < 4; ++i) {
        #pragma unroll
        for (int j = 0; j < 2; ++j) {
            int col = bn + wcol + j * 16 + l16;
            float bv = bias[col];
            #pragma unroll
            for (int v = 0; v < 8; ++v) {
                int row = bm + wrow + i * 16 + v + 8 * lh;
                float val = acc[i][j][v] + bv;
                if (EPI == 0) {
                    if (col < qcols) val *= qscale;
                    ((half_t*)out)[(size_t)row * Nn + col] = (half_t)val;
                } else if (EPI == 1) {
                    ((half_t*)out)[(size_t)row * Nn + col] = (half_t)gelu_exact(val);
                } else if (EPI == 2) {
                    resid[(size_t)row * Nn + col] += val;
                } else {
                    ((float*)out)[(size_t)row * Nn + col] = val;
                }
            }
        }
    }
}

// ---------------------------------------------------------------------------
// Windowed attention: one block per (window, head). N=128, HD=32.
// qkv: [NW*128, 768] f16 (q pre-scaled); outbuf: [NW*128, 256] f16
// MASKED: branch-free shift-mask from LDS-staged group ids
// ---------------------------------------------------------------------------
#define ATTN_SMEM (128*129*4 + 128*136*2 + 32*136*2 + 128)

template <bool MASKED>
__global__ __launch_bounds__(256) void k_attn(
    const half_t* __restrict__ qkv, const float* __restrict__ biasTab,
    const unsigned char* __restrict__ grp, half_t* __restrict__ outbuf)
{
    extern __shared__ char smem[];
    float*  S  = (float*)smem;                                  // [128][129]
    half_t* P  = (half_t*)(smem + 128 * 129 * 4);               // [128][136]
    half_t* vT = (half_t*)(smem + 128*129*4 + 128*136*2);       // [32][136]
    unsigned char* gsh = (unsigned char*)(smem + 128*129*4 + 128*136*2 + 32*136*2);

    int bid = blockIdx.x;
    int win = bid >> 3, h = bid & 7;
    int tid = threadIdx.x, wave = tid >> 5, lane = tid & 31;
    int l16 = lane & 15, lh = lane >> 4;

    const half_t* qbase = qkv + (size_t)win * 128 * 768 + h * 32;
    const half_t* kbase = qbase + 256;
    const half_t* vbase = qbase + 512;

    // stage v^T into LDS: vT[d][m]
    {
        int m = tid >> 1, d0 = (tid & 1) * 16;
        v8h x0 = *(const v8h*)(vbase + (size_t)m * 768 + d0);
        v8h x1 = *(const v8h*)(vbase + (size_t)m * 768 + d0 + 8);
        #pragma unroll
        for (int j = 0; j < 8; ++j) vT[(d0 + j) * 136 + m]     = x0[j];
        #pragma unroll
        for (int j = 0; j < 8; ++j) vT[(d0 + 8 + j) * 136 + m] = x1[j];
    }
    // stage shift-mask group ids (128 bytes) once per block
    if (MASKED && tid < 128) gsh[tid] = grp[(size_t)(win & 255) * 128 + tid];
    if (MASKED) __syncthreads();

    // S = q * k^T for key columns [wave*16, wave*16+16)
    {
        int mcol = wave * 16;
        const half_t* kp = kbase + (size_t)(mcol + l16) * 768 + lh * 8;
        v16h bf = ldfrag(kp, kp + 16);
        const float* bt = biasTab + (size_t)h * 128 * 128;
        unsigned char gc = 0;
        if (MASKED) gc = gsh[mcol + l16];      // column group: fixed per lane
        #pragma unroll
        for (int i = 0; i < 8; ++i) {
            const half_t* qp = qbase + (size_t)(i * 16 + l16) * 768 + lh * 8;
            v16h af = ldfrag(qp, qp + 16);
            v8f c = {};
            c = wmma_f16(af, bf, c);
            #pragma unroll
            for (int v = 0; v < 8; ++v) {
                int row = i * 16 + v + 8 * lh;
                int col = mcol + l16;
                float val = c[v] + bt[row * 128 + col];
                if (MASKED) val += (gsh[row] != gc) ? -100.0f : 0.0f;
                S[row * 129 + col] = val;
            }
        }
    }
    __syncthreads();

    // row softmax (128 rows, one thread each)
    if (tid < 128) {
        float* sr = S + tid * 129;
        float mx = -1e30f;
        for (int j = 0; j < 128; ++j) mx = fmaxf(mx, sr[j]);
        float sum = 0.f;
        for (int j = 0; j < 128; ++j) { float e = __expf(sr[j] - mx); sr[j] = e; sum += e; }
        float inv = 1.0f / sum;
        half_t* pr = P + tid * 136;
        for (int j = 0; j < 128; ++j) pr[j] = (half_t)(sr[j] * inv);
    }
    __syncthreads();

    // O = P * V : wave handles rows [wave*16, +16), cols 0..31 (2 tiles), K=128
    {
        v8f acc[2] = {};
        #pragma unroll
        for (int ks = 0; ks < 128; ks += 32) {
            const half_t* ap = P + (wave * 16 + l16) * 136 + ks + lh * 8;
            v16h af = ldfrag(ap, ap + 16);
            #pragma unroll
            for (int j = 0; j < 2; ++j) {
                const half_t* bp = vT + (j * 16 + l16) * 136 + ks + lh * 8;
                v16h bf = ldfrag(bp, bp + 16);
                acc[j] = wmma_f16(af, bf, acc[j]);
            }
        }
        half_t* ob = outbuf + (size_t)win * 128 * 256 + h * 32;
        #pragma unroll
        for (int j = 0; j < 2; ++j)
            #pragma unroll
            for (int v = 0; v < 8; ++v) {
                int row = wave * 16 + v + 8 * lh;
                int d   = j * 16 + l16;
                ob[(size_t)row * 256 + d] = (half_t)acc[j][v];
            }
    }
}

// ---------------------------------------------------------------------------
// window-reverse (+ roll back) residual scatter-add. grid = 131072 tokens,
// 256 threads = channels
// ---------------------------------------------------------------------------
__global__ __launch_bounds__(256) void k_scatter_add(
    const float* __restrict__ src, float* __restrict__ x, int st, int sh, int sw)
{
    int p = blockIdx.x, c = threadIdx.x;
    int b   = p >> 15;
    int rem = p & 32767;
    int t   = rem >> 12;
    int hw  = rem & 4095;
    int hh  = hw >> 6, wc = hw & 63;
    int qt = (t - st) & 7, qh = (hh - sh) & 63, qw = (wc - sw) & 63;
    int wt = qt >> 1, lt = qt & 1;
    int wh = qh >> 3, lh = qh & 7;
    int ww = qw >> 3, lw = qw & 7;
    long win = (long)b * 256 + wt * 64 + wh * 8 + ww;
    int  n   = lt * 64 + lh * 8 + lw;
    x[(long)p * 256 + c] += src[(win * 128 + n) * 256 + c];
}

// ---------------------------------------------------------------------------
// launch
// ---------------------------------------------------------------------------
extern "C" void kernel_launch(void* const* d_in, const int* in_sizes, int n_in,
                              void* d_out, int out_size, void* d_ws, size_t ws_size,
                              hipStream_t stream) {
    (void)in_sizes; (void)n_in; (void)out_size; (void)ws_size;
    constexpr long NTOK = 131072;          // 4*8*64*64
    const float SCALE = 0.17677669529663687f;   // 32^-0.5

    const float* x       = (const float*)d_in[0];
    const float* rpb1    = (const float*)d_in[1];
    const float* qkv_w1  = (const float*)d_in[2];
    const float* qkv_b1  = (const float*)d_in[3];
    const float* proj_w1 = (const float*)d_in[4];
    const float* proj_b1 = (const float*)d_in[5];
    const float* rpb2    = (const float*)d_in[6];
    const float* qkv_w2  = (const float*)d_in[7];
    const float* qkv_b2  = (const float*)d_in[8];
    const float* proj_w2 = (const float*)d_in[9];
    const float* proj_b2 = (const float*)d_in[10];
    const float* g1 = (const float*)d_in[11], *be1 = (const float*)d_in[12];
    const float* g2 = (const float*)d_in[13], *be2 = (const float*)d_in[14];
    const float* g3 = (const float*)d_in[15], *be3 = (const float*)d_in[16];
    const float* g4 = (const float*)d_in[17], *be4 = (const float*)d_in[18];
    const float* m1w1 = (const float*)d_in[19], *m1b1 = (const float*)d_in[20];
    const float* m1w2 = (const float*)d_in[21], *m1b2 = (const float*)d_in[22];
    const float* m2w1 = (const float*)d_in[23], *m2b1 = (const float*)d_in[24];
    const float* m2w2 = (const float*)d_in[25], *m2b2 = (const float*)d_in[26];

    char* wsp = (char*)d_ws;
    size_t off = 0;
    auto alloc = [&](size_t bytes) -> char* {
        char* p = wsp + off;
        off += (bytes + 255) & ~(size_t)255;
        return p;
    };
    half_t* w_qkv1 = (half_t*)alloc(196608 * 2);
    half_t* w_prj1 = (half_t*)alloc(65536 * 2);
    half_t* w_m1a  = (half_t*)alloc(131072 * 2);
    half_t* w_m1b  = (half_t*)alloc(131072 * 2);
    half_t* w_qkv2 = (half_t*)alloc(196608 * 2);
    half_t* w_prj2 = (half_t*)alloc(65536 * 2);
    half_t* w_m2a  = (half_t*)alloc(131072 * 2);
    half_t* w_m2b  = (half_t*)alloc(131072 * 2);
    float*  bt1    = (float*)alloc(8 * 128 * 128 * 4);
    float*  bt2    = (float*)alloc(8 * 128 * 128 * 4);
    unsigned char* grp = (unsigned char*)alloc(256 * 128);
    half_t* bufB   = (half_t*)alloc(NTOK * 256 * 2);   // LN / windowed activations
    half_t* bufC   = (half_t*)alloc(NTOK * 256 * 2);   // attention output
    char*   bufA   = alloc(NTOK * 768 * 2);            // qkv f16 / proj f32 / mlp hidden
    half_t* qkvbuf  = (half_t*)bufA;
    float*  projtmp = (float*)bufA;
    half_t* hid     = (half_t*)bufA;
    float*  xout    = (float*)d_out;

    // setup: weights -> f16, bias tables, shift groups, x -> d_out
    auto cvt = [&](const float* a, half_t* b, long n) {
        k_f32_to_f16<<<(int)((n + 255) / 256), 256, 0, stream>>>(a, b, n);
    };
    cvt(qkv_w1, w_qkv1, 196608); cvt(proj_w1, w_prj1, 65536);
    cvt(m1w1, w_m1a, 131072);    cvt(m1w2, w_m1b, 131072);
    cvt(qkv_w2, w_qkv2, 196608); cvt(proj_w2, w_prj2, 65536);
    cvt(m2w1, w_m2a, 131072);    cvt(m2w2, w_m2b, 131072);
    k_build_bias<<<1024, 128, 0, stream>>>(rpb1, bt1);
    k_build_bias<<<1024, 128, 0, stream>>>(rpb2, bt2);
    k_build_grp<<<256, 128, 0, stream>>>(grp);
    hipMemcpyAsync(d_out, (const void*)x, (size_t)NTOK * 256 * 4,
                   hipMemcpyDeviceToDevice, stream);

    // ---- block 1 (no shift, no mask) ----
    k_layernorm<<<16384, 256, 0, stream>>>(xout, g1, be1, bufB, 1, 0, 0, 0);
    k_gemm<0><<<dim3(6, 1024), 256, 0, stream>>>(bufB, w_qkv1, qkv_b1, qkvbuf,
                                                 nullptr, 131072, 768, 256, SCALE, 256);
    k_attn<false><<<8192, 256, ATTN_SMEM, stream>>>(qkvbuf, bt1, nullptr, bufC);
    k_gemm<3><<<dim3(2, 1024), 256, 0, stream>>>(bufC, w_prj1, proj_b1, projtmp,
                                                 nullptr, 131072, 256, 256, 0.f, 0);
    k_scatter_add<<<131072, 256, 0, stream>>>(projtmp, xout, 0, 0, 0);
    k_layernorm<<<16384, 256, 0, stream>>>(xout, g2, be2, bufB, 0, 0, 0, 0);
    k_gemm<1><<<dim3(4, 1024), 256, 0, stream>>>(bufB, w_m1a, m1b1, hid,
                                                 nullptr, 131072, 512, 256, 0.f, 0);
    k_gemm<2><<<dim3(2, 1024), 256, 0, stream>>>(hid, w_m1b, m1b2, nullptr,
                                                 xout, 131072, 256, 512, 0.f, 0);

    // ---- block 2 (shift (1,4,4), mask) ----
    k_layernorm<<<16384, 256, 0, stream>>>(xout, g3, be3, bufB, 1, 1, 4, 4);
    k_gemm<0><<<dim3(6, 1024), 256, 0, stream>>>(bufB, w_qkv2, qkv_b2, qkvbuf,
                                                 nullptr, 131072, 768, 256, SCALE, 256);
    k_attn<true><<<8192, 256, ATTN_SMEM, stream>>>(qkvbuf, bt2, grp, bufC);
    k_gemm<3><<<dim3(2, 1024), 256, 0, stream>>>(bufC, w_prj2, proj_b2, projtmp,
                                                 nullptr, 131072, 256, 256, 0.f, 0);
    k_scatter_add<<<131072, 256, 0, stream>>>(projtmp, xout, 1, 4, 4);
    k_layernorm<<<16384, 256, 0, stream>>>(xout, g4, be4, bufB, 0, 0, 0, 0);
    k_gemm<1><<<dim3(4, 1024), 256, 0, stream>>>(bufB, w_m2a, m2b1, hid,
                                                 nullptr, 131072, 512, 256, 0.f, 0);
    k_gemm<2><<<dim3(2, 1024), 256, 0, stream>>>(hid, w_m2b, m2b2, nullptr,
                                                 xout, 131072, 256, 512, 0.f, 0);
}
